// ConvEGNN4_44822278701380
// MI455X (gfx1250) — matrix operand order repaired and checked by reading, not compiled, not run on hardware
//
#include <hip/hip_runtime.h>
#include <hip/hip_bf16.h>
#include <math.h>

typedef __attribute__((ext_vector_type(16))) _Float16 v16h;
typedef __attribute__((ext_vector_type(8)))  _Float16 v8h;
typedef __attribute__((ext_vector_type(8)))  float    v8f;

#define HID   64
#define KNB   32
#define NNODE (16 * 1024)
#define K1PAD 160   // 64 + 64 + dist + speed + 1(bias) + pad   (also split h-MLP l1)
#define K2PAD 96    // 64 + 1(bias) + pad
#define E1STR 168   // LDS row stride (halves), 336B = 16B multiple
#define E2STR 104   // LDS row stride (halves), 208B = 16B multiple

__device__ __forceinline__ float silu_f(float x) { return x / (1.f + __expf(-x)); }

// ---------------------------------------------------------------------------
// Prep: f16 transposed, K-padded weight [64][Kpad]; row K holds the bias.
// W row-major [K][64] (jax (fan_in, fan_out)); dst[n][kk].
// ---------------------------------------------------------------------------
__global__ void prep_wt(const float* __restrict__ W, const float* __restrict__ b,
                        _Float16* __restrict__ dst, int K, int Kpad) {
    int idx = blockIdx.x * blockDim.x + threadIdx.x;
    if (idx >= HID * Kpad) return;
    int n = idx / Kpad, kk = idx % Kpad;
    float v = 0.f;
    if (kk < K)       v = W[kk * HID + n];
    else if (kk == K) v = b[n];
    dst[n * Kpad + kk] = (_Float16)v;
}

// Split layout for W_h1 [128][64]: A-matrix is [h_st 0..63 | 1 @64 | 0 65..95 |
// m_i 96..159], so dst kk<64 -> W rows 0..63; kk==64 -> bias; 65..95 -> 0;
// kk 96..159 -> W rows 64..127.
__global__ void prep_wh1(const float* __restrict__ W, const float* __restrict__ b,
                         _Float16* __restrict__ dst) {
    int idx = blockIdx.x * blockDim.x + threadIdx.x;
    if (idx >= HID * K1PAD) return;
    int n = idx / K1PAD, kk = idx % K1PAD;
    float v = 0.f;
    if (kk < 64)       v = W[kk * HID + n];
    else if (kk == 64) v = b[n];
    else if (kk >= 96) v = W[(kk - 32) * HID + n];
    dst[n * K1PAD + kk] = (_Float16)v;
}

// ---------------------------------------------------------------------------
// One wave: 32xKpad (LDS f16) @ Kpadx16 (global f16, transposed) -> two 16x16
// f32 accumulators (rows 0..15, 16..31); N-tile = jtile.
// ---------------------------------------------------------------------------
template<int STRIDE, int KPAD, int KCHUNKS>
__device__ __forceinline__ void wmma_tile(const _Float16* __restrict__ Eb,
                                          const _Float16* __restrict__ Wt,
                                          int jtile, int lane,
                                          v8f& acc0, v8f& acc1) {
    const int rlo = lane & 15;
    const int hiA = (lane & 16) ? 8 : 0;
    const int hiB = (lane & 16) ? 8 : 0;
    const _Float16* a0p = Eb + rlo * STRIDE + hiA;
    const _Float16* a1p = a0p + 16 * STRIDE;
    const _Float16* bp  = Wt + (jtile * 16 + rlo) * KPAD + hiB;
#pragma unroll
    for (int c = 0; c < KCHUNKS; ++c) {
        const int kb = c * 32;
        v8h b0  = *(const v8h*)(bp  + kb);
        v8h b1  = *(const v8h*)(bp  + kb + 16);
        v8h al0 = *(const v8h*)(a0p + kb);
        v8h ah0 = *(const v8h*)(a0p + kb + 16);
        v8h al1 = *(const v8h*)(a1p + kb);
        v8h ah1 = *(const v8h*)(a1p + kb + 16);
        v16h B, A0, A1;
#pragma unroll
        for (int i = 0; i < 8; ++i) {
            B[i]  = b0[i];  B[i + 8]  = b1[i];
            A0[i] = al0[i]; A0[i + 8] = ah0[i];
            A1[i] = al1[i]; A1[i + 8] = ah1[i];
        }
        acc0 = __builtin_amdgcn_wmma_f32_16x16x32_f16(false, A0, false, B,
                                                      (short)0, acc0, false, false);
        acc1 = __builtin_amdgcn_wmma_f32_16x16x32_f16(false, A1, false, B,
                                                      (short)0, acc1, false, false);
    }
}

// ---------------------------------------------------------------------------
// Kernel 1: edge pipeline. One block (128 thr = 4 wave32) per node.
// Exports m_i[64] (f32) and agg[2] (pre-divided) per node to workspace.
// ---------------------------------------------------------------------------
__global__ __launch_bounds__(128)
void egnn_edge_kernel(const float* __restrict__ hst_g, const float* __restrict__ hneigh,
                      const float* __restrict__ rela,  const int* __restrict__ nmask,
                      const _Float16* __restrict__ W1t, const _Float16* __restrict__ W2t,
                      const _Float16* __restrict__ Wx1t,
                      const float* __restrict__ Wx2, const float* __restrict__ bx2,
                      float* __restrict__ MiW, float* __restrict__ AggW) {
    __shared__ __align__(16) _Float16 E1[KNB][E1STR];
    __shared__ __align__(16) _Float16 E2[KNB][E2STR];
    __shared__ __align__(16) _Float16 E3[KNB][E2STR];
    __shared__ float mi[HID];
    __shared__ float scal[KNB];
    __shared__ float maskf[KNB];

    const int bn   = blockIdx.x;
    const int t    = threadIdx.x;
    const int lane = t & 31;
    const int jt   = t >> 5;

    const float* hst = hst_g  + (size_t)bn * HID;
    const float* hn  = hneigh + (size_t)bn * KNB * HID;
    const float* rel = rela   + (size_t)bn * KNB * 6;
    const int*   msk = nmask  + (size_t)bn * KNB;

    if (t == 0) {
        __builtin_prefetch(W1t, 0, 1);
        __builtin_prefetch(W2t, 0, 1);
        __builtin_prefetch(Wx1t, 0, 1);
    }

    // ---- Stage A: stage edge-feature matrix + constants ----
    for (int i = t; i < KNB * HID; i += 128) {
        const int r = i >> 6, c = i & 63;
        E1[r][c]      = (_Float16)hst[c];
        E1[r][64 + c] = (_Float16)hn[i];
    }
    for (int i = t; i < HID; i += 128) mi[i] = 0.f;
    if (t < KNB) {
        const float rx = rel[t * 6 + 0], ry = rel[t * 6 + 1];
        const float vx = rel[t * 6 + 2], vy = rel[t * 6 + 3];
        E1[t][128] = (_Float16)sqrtf(rx * rx + ry * ry);
        E1[t][129] = (_Float16)sqrtf(vx * vx + vy * vy);
        E1[t][130] = (_Float16)1.f;
#pragma unroll
        for (int c2 = 131; c2 < K1PAD; ++c2) E1[t][c2] = (_Float16)0.f;
        maskf[t] = msk[t] ? 1.f : 0.f;
        scal[t]  = bx2[0];
    }
    for (int i = t; i < KNB * 32; i += 128) {
        const int r = i >> 5, c = 64 + (i & 31);
        const _Float16 v = (c == 64) ? (_Float16)1.f : (_Float16)0.f;
        E2[r][c] = v; E3[r][c] = v;
    }
    __syncthreads();

    // ---- e-MLP layer 1 (K=160): silu -> E2 ----
    v8f acc0, acc1;
#pragma unroll
    for (int i = 0; i < 8; ++i) { acc0[i] = 0.f; acc1[i] = 0.f; }
    wmma_tile<E1STR, K1PAD, 5>(&E1[0][0], W1t, jt, lane, acc0, acc1);
    {
        const int col = jt * 16 + (lane & 15);
        const int rb  = (lane & 16) ? 8 : 0;
#pragma unroll
        for (int v = 0; v < 8; ++v) {
            E2[rb + v][col]      = (_Float16)silu_f(acc0[v]);
            E2[rb + v + 16][col] = (_Float16)silu_f(acc1[v]);
        }
    }
    __syncthreads();

    // ---- e-MLP layer 2 (K=96): silu, mask -> E3 ; accumulate m_i ----
#pragma unroll
    for (int i = 0; i < 8; ++i) { acc0[i] = 0.f; acc1[i] = 0.f; }
    wmma_tile<E2STR, K2PAD, 3>(&E2[0][0], W2t, jt, lane, acc0, acc1);
    {
        const int col = jt * 16 + (lane & 15);
        const int rb  = (lane & 16) ? 8 : 0;
        float part = 0.f;
#pragma unroll
        for (int v = 0; v < 8; ++v) {
            const float m0 = maskf[rb + v]      * silu_f(acc0[v]);
            const float m1 = maskf[rb + v + 16] * silu_f(acc1[v]);
            E3[rb + v][col]      = (_Float16)m0;
            E3[rb + v + 16][col] = (_Float16)m1;
            part += m0 + m1;
        }
        atomicAdd(&mi[col], part);
    }
    __syncthreads();

    // ---- x-MLP layer 1 (K=96): silu, dot with W_x2 -> per-edge scalar ----
#pragma unroll
    for (int i = 0; i < 8; ++i) { acc0[i] = 0.f; acc1[i] = 0.f; }
    wmma_tile<E2STR, K2PAD, 3>(&E3[0][0], Wx1t, jt, lane, acc0, acc1);
    {
        const int rb = (lane & 16) ? 8 : 0;
        const float wc = Wx2[jt * 16 + (lane & 15)];
#pragma unroll
        for (int v = 0; v < 8; ++v) {
            atomicAdd(&scal[rb + v],      silu_f(acc0[v]) * wc);
            atomicAdd(&scal[rb + v + 16], silu_f(acc1[v]) * wc);
        }
    }
    __syncthreads();

    // ---- Export: agg (wave0 shfl reduce) and m_i ----
    if (t < KNB) {
        const float s = scal[t];
        float sx = rel[t * 6 + 0] * s;
        float sy = rel[t * 6 + 1] * s;
        float nn = maskf[t];
#pragma unroll
        for (int off = 16; off > 0; off >>= 1) {
            sx += __shfl_down(sx, off);
            sy += __shfl_down(sy, off);
            nn += __shfl_down(nn, off);
        }
        if (t == 0) {
            const float inv = 1.f / (nn + 1e-6f);
            AggW[(size_t)bn * 2 + 0] = sx * inv;
            AggW[(size_t)bn * 2 + 1] = sy * inv;
        }
    }
    if (t < HID) MiW[(size_t)bn * HID + t] = mi[t];
}

// ---------------------------------------------------------------------------
// Kernel 2: node pipeline, 32 nodes per block via WMMA.
// A1 row = node: [h_st(64) | 1 @64 | 0 (65..95) | m_i (96..159)].
//   a-MLP layer1 uses K-chunks 0..2 (Wa1t, Kpad=96)
//   h-MLP layer1 uses K-chunks 0..4 (Wh1t split layout, Kpad=160)
// ---------------------------------------------------------------------------
__global__ __launch_bounds__(128)
void egnn_node_kernel(const float* __restrict__ ped, const float* __restrict__ hst_g,
                      const float* __restrict__ MiW, const float* __restrict__ AggW,
                      const _Float16* __restrict__ Wa1t, const _Float16* __restrict__ Wh1t,
                      const _Float16* __restrict__ Wh2t,
                      const float* __restrict__ Wa2, const float* __restrict__ ba2,
                      float* __restrict__ out1, float* __restrict__ outh) {
    __shared__ __align__(16) _Float16 A1[KNB][E1STR];
    __shared__ __align__(16) _Float16 G[KNB][E2STR];
    __shared__ float sa[KNB];

    const int nb   = blockIdx.x;      // tile of 32 nodes
    const int t    = threadIdx.x;
    const int lane = t & 31;
    const int jt   = t >> 5;
    const size_t base = (size_t)nb * KNB;

    // ---- Stage A ----
    for (int i = t; i < KNB * HID; i += 128) {
        const int r = i >> 6, c = i & 63;
        A1[r][c]      = (_Float16)hst_g[(base + r) * HID + c];
        A1[r][96 + c] = (_Float16)MiW[(base + r) * HID + c];
    }
    if (t < KNB) {
        A1[t][64] = (_Float16)1.f;
#pragma unroll
        for (int c2 = 65; c2 < 96; ++c2) A1[t][c2] = (_Float16)0.f;
        sa[t] = ba2[0];
    }
    for (int i = t; i < KNB * 32; i += 128) {
        const int r = i >> 5, c = 64 + (i & 31);
        G[r][c] = (c == 64) ? (_Float16)1.f : (_Float16)0.f;
    }
    __syncthreads();

    // ---- h-MLP layer 1 (K=160 split): silu -> G ----
    v8f acc0, acc1;
#pragma unroll
    for (int i = 0; i < 8; ++i) { acc0[i] = 0.f; acc1[i] = 0.f; }
    wmma_tile<E1STR, K1PAD, 5>(&A1[0][0], Wh1t, jt, lane, acc0, acc1);
    {
        const int col = jt * 16 + (lane & 15);
        const int rb  = (lane & 16) ? 8 : 0;
#pragma unroll
        for (int v = 0; v < 8; ++v) {
            G[rb + v][col]      = (_Float16)silu_f(acc0[v]);
            G[rb + v + 16][col] = (_Float16)silu_f(acc1[v]);
        }
    }

    // ---- a-MLP layer 1 (K=96): silu, dot with W_a2 -> per-node scalar ----
#pragma unroll
    for (int i = 0; i < 8; ++i) { acc0[i] = 0.f; acc1[i] = 0.f; }
    wmma_tile<E1STR, K2PAD, 3>(&A1[0][0], Wa1t, jt, lane, acc0, acc1);
    {
        const int rb = (lane & 16) ? 8 : 0;
        const float wc = Wa2[jt * 16 + (lane & 15)];
#pragma unroll
        for (int v = 0; v < 8; ++v) {
            atomicAdd(&sa[rb + v],      silu_f(acc0[v]) * wc);
            atomicAdd(&sa[rb + v + 16], silu_f(acc1[v]) * wc);
        }
    }
    __syncthreads();

    // ---- h-MLP layer 2 (K=96): h_new = h_st + (G @ W_h2 + b_h2) ----
#pragma unroll
    for (int i = 0; i < 8; ++i) { acc0[i] = 0.f; acc1[i] = 0.f; }
    wmma_tile<E2STR, K2PAD, 3>(&G[0][0], Wh2t, jt, lane, acc0, acc1);
    {
        const int col = jt * 16 + (lane & 15);
        const int rb  = (lane & 16) ? 8 : 0;
#pragma unroll
        for (int v = 0; v < 8; ++v) {
            const size_t i0 = (base + rb + v) * HID + col;
            const size_t i1 = (base + rb + v + 16) * HID + col;
            outh[i0] = hst_g[i0] + acc0[v];
            outh[i1] = hst_g[i1] + acc1[v];
        }
    }

    // ---- x/v/a outputs ----
    if (t < KNB) {
        const size_t g = base + t;
        const float aggx = AggW[g * 2 + 0];
        const float aggy = AggW[g * 2 + 1];
        const float* pf  = ped + g * 6;
        const float av   = sa[t];
        const float ax = av * pf[4] + aggx;
        const float ay = av * pf[5] + aggy;
        const float vx = pf[2] + ax;
        const float vy = pf[3] + ay;
        float* o = out1 + g * 6;
        o[0] = pf[0] + vx; o[1] = pf[1] + vy;
        o[2] = vx;         o[3] = vy;
        o[4] = ax;         o[5] = ay;
    }
}

// ---------------------------------------------------------------------------
extern "C" void kernel_launch(void* const* d_in, const int* in_sizes, int n_in,
                              void* d_out, int out_size, void* d_ws, size_t ws_size,
                              hipStream_t stream) {
    (void)in_sizes; (void)n_in; (void)out_size; (void)ws_size;
    const float* ped  = (const float*)d_in[0];
    const float* hst  = (const float*)d_in[1];
    const float* hngh = (const float*)d_in[2];
    const float* rela = (const float*)d_in[3];
    const int*   msk  = (const int*)d_in[4];
    const float* We1 = (const float*)d_in[5];  const float* be1 = (const float*)d_in[6];
    const float* We2 = (const float*)d_in[7];  const float* be2 = (const float*)d_in[8];
    const float* Wx1 = (const float*)d_in[9];  const float* bx1 = (const float*)d_in[10];
    const float* Wx2 = (const float*)d_in[11]; const float* bx2 = (const float*)d_in[12];
    const float* Wa1 = (const float*)d_in[13]; const float* ba1 = (const float*)d_in[14];
    const float* Wa2 = (const float*)d_in[15]; const float* ba2 = (const float*)d_in[16];
    const float* Wh1 = (const float*)d_in[17]; const float* bh1 = (const float*)d_in[18];
    const float* Wh2 = (const float*)d_in[19]; (void)d_in[20];

    // workspace layout (f16 weights, then f32 Mi / Agg; all 16B aligned)
    _Float16* w1t  = (_Float16*)d_ws;              // [64][160]
    _Float16* w2t  = w1t  + HID * K1PAD;           // [64][96]
    _Float16* wx1t = w2t  + HID * K2PAD;           // [64][96]
    _Float16* wa1t = wx1t + HID * K2PAD;           // [64][96]
    _Float16* wh1t = wa1t + HID * K2PAD;           // [64][160] split layout
    _Float16* wh2t = wh1t + HID * K1PAD;           // [64][96]
    float* MiW  = (float*)(wh2t + HID * K2PAD);    // [16384][64]
    float* AggW = MiW + (size_t)NNODE * HID;       // [16384][2]

    prep_wt <<<(HID * K1PAD + 255) / 256, 256, 0, stream>>>(We1, be1, w1t, 130, K1PAD);
    prep_wt <<<(HID * K2PAD + 255) / 256, 256, 0, stream>>>(We2, be2, w2t, HID, K2PAD);
    prep_wt <<<(HID * K2PAD + 255) / 256, 256, 0, stream>>>(Wx1, bx1, wx1t, HID, K2PAD);
    prep_wt <<<(HID * K2PAD + 255) / 256, 256, 0, stream>>>(Wa1, ba1, wa1t, HID, K2PAD);
    prep_wh1<<<(HID * K1PAD + 255) / 256, 256, 0, stream>>>(Wh1, bh1, wh1t);
    prep_wt <<<(HID * K2PAD + 255) / 256, 256, 0, stream>>>(Wh2, d_in[20] ? (const float*)d_in[20] : Wh2, wh2t, HID, K2PAD);

    egnn_edge_kernel<<<NNODE, 128, 0, stream>>>(
        hst, hngh, rela, msk, w1t, w2t, wx1t, Wx2, bx2, MiW, AggW);

    float* out1 = (float*)d_out;
    float* outh = out1 + (size_t)NNODE * 6;
    egnn_node_kernel<<<NNODE / KNB, 128, 0, stream>>>(
        ped, hst, MiW, AggW, wa1t, wh1t, wh2t, Wa2, ba2, out1, outh);
}
